// MultiHeadAttention_64716567216712
// MI455X (gfx1250) — matrix operand order repaired and checked by reading, not compiled, not run on hardware
//
#include <hip/hip_runtime.h>
#include <hip/hip_bf16.h>
#include <math.h>

typedef __attribute__((ext_vector_type(16))) __bf16 vbf16x16;
typedef __attribute__((ext_vector_type(8)))  __bf16 vbf16x8;
typedef __attribute__((ext_vector_type(4)))  __bf16 vbf16x4;
typedef __attribute__((ext_vector_type(8)))  float  vf32x8;
typedef __attribute__((ext_vector_type(4)))  unsigned int ui32x4;
typedef __attribute__((ext_vector_type(8)))  int    i32x8;
typedef __attribute__((ext_vector_type(4)))  int    i32x4;

#define B_   2
#define S_   2048
#define E_   1024
#define H_   16
#define DK_  64
#define M_   (B_ * S_)

#define TBM  256        // block rows (8 waves x 32 rows)
#define TBN  64         // block cols
#define TBK  32

#if __has_builtin(__builtin_amdgcn_tensor_load_to_lds) && __has_builtin(__builtin_amdgcn_s_wait_tensorcnt)
#define HAVE_TDM 1
#else
#define HAVE_TDM 0
#endif

__device__ __forceinline__ vbf16x16 cat8(vbf16x8 a, vbf16x8 b) {
    return __builtin_shufflevector(a, b, 0,1,2,3,4,5,6,7,8,9,10,11,12,13,14,15);
}

#if HAVE_TDM
// 2D TDM load: tile (tile_d0 x tile_d1) of bf16 from row-major tensor with row
// stride stride0 (elements) into LDS at lds_off, optional LDS row padding.
__device__ __forceinline__ void tdm_load_2d(unsigned lds_off, const void* gaddr,
                                            unsigned tile_d0, unsigned tile_d1,
                                            unsigned tensor_d0, unsigned tensor_d1,
                                            unsigned long long stride0,
                                            unsigned pad_en, unsigned pad_int_code,
                                            unsigned pad_amt_code)
{
    unsigned long long ga = (unsigned long long)(uintptr_t)gaddr;
    ui32x4 g0;
    g0[0] = 1u;                                              // count=1, user mode
    g0[1] = lds_off;                                         // lds_addr (bytes)
    g0[2] = (unsigned)(ga & 0xFFFFFFFFu);                    // global_addr[31:0]
    g0[3] = (unsigned)((ga >> 32) & 0x01FFFFFFu) | (2u << 30); // addr[56:32] | type=2
    i32x8 g1;
    g1[0] = (int)((1u << 16) | (pad_en << 20) |
                  (pad_int_code << 22) | (pad_amt_code << 25)); // data_size=1 (2B)
    g1[1] = (int)((tensor_d0 & 0xFFFFu) << 16);              // tensor_dim0[15:0]
    g1[2] = (int)(((tensor_d0 >> 16) & 0xFFFFu) | ((tensor_d1 & 0xFFFFu) << 16));
    g1[3] = (int)(((tensor_d1 >> 16) & 0xFFFFu) | (tile_d0 << 16));
    g1[4] = (int)(tile_d1 & 0xFFFFu);                        // tile_dim2 = 0
    g1[5] = (int)(stride0 & 0xFFFFFFFFu);                    // tensor_dim0_stride lo
    g1[6] = (int)((stride0 >> 32) & 0xFFFFu);                // stride hi | dim1_stride=0
    g1[7] = 0;
    i32x4 z4 = {};
#if __clang_major__ >= 23
    i32x8 z8 = {};
    __builtin_amdgcn_tensor_load_to_lds(g0, g1, z4, z4, z8, 0);
#else
    __builtin_amdgcn_tensor_load_to_lds(g0, g1, z4, z4, 0);
#endif
}
#endif

// ---------------------------------------------------------------------------
// Kernel 1: QKV projection (+RoPE on Q,K). fp32 in -> bf16 out.
// Q,K layout [B,H,S,DK]; V stored transposed [B,H,DK,S].
// Each wave computes a 32x64 output tile (2 A-frags reuse 4 B-frags).
// ---------------------------------------------------------------------------
__global__ __launch_bounds__(256) void qkv_rope_kernel(
    const float* __restrict__ x,  const float* __restrict__ wq,
    const float* __restrict__ wk, const float* __restrict__ wv,
    __bf16* __restrict__ qo, __bf16* __restrict__ ko, __bf16* __restrict__ vo)
{
    __shared__ alignas(16) __bf16 Xs[TBM][TBK + 8];      // row-major A tile
    __shared__ alignas(32) __bf16 WsB[4][32][16];        // B-fragment swizzled

    const int tid  = threadIdx.x;
    const int w    = tid >> 5;
    const int lane = tid & 31;
    const int hi   = lane >> 4;
    const int col  = lane & 15;
    const int m0   = blockIdx.x * TBM;
    const int n0   = blockIdx.y * TBN;
    const int z    = blockIdx.z;
    const float* wmat = (z == 0) ? wq : (z == 1) ? wk : wv;

    const int nloc = tid & 63;            // W loader: local n
    const int kg   = tid >> 6;            // W loader: k-group of 8

    vf32x8 acc[2][4] = {};

    for (int k0 = 0; k0 < E_; k0 += TBK) {
        __syncthreads();
        #pragma unroll
        for (int i = 0; i < 8; ++i) {                    // X tile: float4 -> 4xbf16
            int idx = i * 256 + tid;
            int r = idx >> 3, c4 = idx & 7;
            float4 f = ((const float4*)(x + (size_t)(m0 + r) * E_ + k0))[c4];
            vbf16x4 h4 = { (__bf16)f.x, (__bf16)f.y, (__bf16)f.z, (__bf16)f.w };
            *(vbf16x4*)&Xs[r][c4 * 4] = h4;
        }
        {                                                // W tile, swizzled store
            vbf16x8 c8;
            #pragma unroll
            for (int i2 = 0; i2 < 8; ++i2)
                c8[i2] = (__bf16)wmat[(size_t)(k0 + kg * 8 + i2) * E_ + n0 + nloc];
            int dl = (nloc & 15) | ((kg >> 1) << 4);
            *(vbf16x8*)&WsB[nloc >> 4][dl][(kg & 1) * 8] = c8;
        }
        __syncthreads();

        const vbf16x8* xp0 = (const vbf16x8*)&Xs[w * 32 + col][0];
        const vbf16x8* xp1 = (const vbf16x8*)&Xs[w * 32 + 16 + col][0];
        vbf16x16 a0 = cat8(xp0[2 * hi], xp0[2 * hi + 1]);
        vbf16x16 a1 = cat8(xp1[2 * hi], xp1[2 * hi + 1]);
        #pragma unroll
        for (int nc = 0; nc < 4; ++nc) {
            vbf16x16 bfrag = *(const vbf16x16*)&WsB[nc][lane][0];
            acc[0][nc] = __builtin_amdgcn_wmma_f32_16x16x32_bf16(
                false, a0, false, bfrag, (short)0, acc[0][nc], false, false);
            acc[1][nc] = __builtin_amdgcn_wmma_f32_16x16x32_bf16(
                false, a1, false, bfrag, (short)0, acc[1][nc], false, false);
        }
    }

    #pragma unroll
    for (int mg = 0; mg < 2; ++mg) {
        if (z == 2) {                                    // V: transposed, vector store
            const int m0w = m0 + w * 32 + mg * 16 + hi * 8;
            const int b = m0w >> 11, s0 = m0w & (S_ - 1);
            #pragma unroll
            for (int nc = 0; nc < 4; ++nc) {
                vbf16x8 pack;
                #pragma unroll
                for (int v = 0; v < 8; ++v) pack[v] = (__bf16)acc[mg][nc][v];
                int n = n0 + nc * 16 + col;
                int h = n >> 6, d = n & 63;
                size_t off = (((size_t)b * H_ + h) * DK_ + d) * S_ + s0;
                *(vbf16x8*)(vo + off) = pack;
            }
        } else {                                         // Q/K: RoPE + scatter
            __bf16* outp = (z == 0) ? qo : ko;
            #pragma unroll
            for (int nc = 0; nc < 4; ++nc) {
                #pragma unroll
                for (int v = 0; v < 8; ++v) {
                    float val = acc[mg][nc][v];
                    int m = m0 + w * 32 + mg * 16 + hi * 8 + v;
                    int n = n0 + nc * 16 + col;
                    int b = m >> 11, s = m & (S_ - 1);
                    int h = n >> 6,  d = n & 63;
                    float partner = __shfl_xor(val, 1, 32);
                    float ang = (float)s * __powf(10000.f, -(float)(d & ~1) * (1.f / 64.f));
                    float sn, cs;
                    __sincosf(ang, &sn, &cs);
                    float rot = (d & 1) ? partner : -partner;
                    val = val * cs + rot * sn;
                    size_t off = (((size_t)b * H_ + h) * S_ + s) * DK_ + d;
                    outp[off] = (__bf16)val;
                }
            }
        }
    }
}

// ---------------------------------------------------------------------------
// Kernel 2: causal flash attention, K/V staged by TDM (double-buffered).
// q,k: [B,H,S,DK]; v: [B,H,DK,S]; out bf16 [B,S,H,DK] == [M,E].
// ---------------------------------------------------------------------------
__global__ __launch_bounds__(256) void attn_kernel(
    const __bf16* __restrict__ q, const __bf16* __restrict__ kmat,
    const __bf16* __restrict__ vt, __bf16* __restrict__ o)
{
    __shared__ alignas(16) __bf16 Ks[2][32][72];         // [key][dk], stride 72
    __shared__ alignas(16) __bf16 Vs[2][64][40];         // [dk][key], stride 40
    __shared__ alignas(16) __bf16 Pl[8][16][40];         // per-wave P transpose

    const int tid  = threadIdx.x;
    const int w    = tid >> 5;
    const int lane = tid & 31;
    const int hi   = lane >> 4;
    const int col  = lane & 15;
    const int bh   = blockIdx.y;
    const int qb   = blockIdx.x * 128 + w * 16;
    const size_t base = (size_t)bh * S_ * DK_;           // q/k base (elements)
    const size_t vbase = (size_t)bh * DK_ * S_;          // v^T base

    // Q fragments (DK=64 -> two K=32 halves), vectorized global loads
    const vbf16x8* qp = (const vbf16x8*)(q + base + (size_t)(qb + col) * DK_);
    vbf16x16 aq[2];
    aq[0] = cat8(qp[hi],     qp[2 + hi]);
    aq[1] = cat8(qp[4 + hi], qp[6 + hi]);

    float mrow[8], lrow[8];
    #pragma unroll
    for (int v = 0; v < 8; ++v) { mrow[v] = -3e38f; lrow[v] = 0.f; }
    vf32x8 acc[4] = {};

    const int nk = blockIdx.x * 4 + 4;

#if HAVE_TDM
    if (w == 0) {                                        // prefetch chunk 0
        tdm_load_2d((unsigned)(uintptr_t)(void*)&Ks[0][0][0], kmat + base,
                    64, 32, 64, S_, 64ull, 1u, 4u, 3u);  // pad 32dw rows by 4dw
        tdm_load_2d((unsigned)(uintptr_t)(void*)&Vs[0][0][0], vt + vbase,
                    32, 64, S_, 64, (unsigned long long)S_, 1u, 3u, 3u);
    }
#endif

    for (int j = 0; j < nk; ++j) {
        const int kstart = j * 32;
        const int cur = j & 1;
        __syncthreads();
#if HAVE_TDM
        if (w == 0) {
            if (j + 1 < nk) {
                const int nx = (j + 1) & 1, ks2 = (j + 1) * 32;
                tdm_load_2d((unsigned)(uintptr_t)(void*)&Ks[nx][0][0],
                            kmat + base + (size_t)ks2 * DK_,
                            64, 32, 64, S_, 64ull, 1u, 4u, 3u);
                tdm_load_2d((unsigned)(uintptr_t)(void*)&Vs[nx][0][0],
                            vt + vbase + ks2,
                            32, 64, S_, 64, (unsigned long long)S_, 1u, 3u, 3u);
                __builtin_amdgcn_s_wait_tensorcnt(2);    // chunk j complete
            } else {
                __builtin_amdgcn_s_wait_tensorcnt(0);
            }
        }
#else
        #pragma unroll
        for (int i = 0; i < 8; ++i) {
            int idx = i * 256 + tid;
            Ks[cur][idx >> 6][idx & 63] =
                kmat[base + (size_t)(kstart + (idx >> 6)) * DK_ + (idx & 63)];
            Vs[cur][idx >> 5][idx & 31] =
                vt[vbase + (size_t)(idx >> 5) * S_ + kstart + (idx & 31)];
        }
#endif
        __syncthreads();
        if (kstart > qb + 15) continue;                  // wave-uniform causal skip

        // ---- scores S = Q K^T / 8, two 16-key sub-chunks ----
        float pv[2][8];
        float chmax[8];
        #pragma unroll
        for (int v = 0; v < 8; ++v) chmax[v] = -3e38f;
        #pragma unroll
        for (int sc = 0; sc < 2; ++sc) {
            vf32x8 sacc = {};
            const vbf16x8* kp = (const vbf16x8*)&Ks[cur][sc * 16 + col][0];
            #pragma unroll
            for (int h2 = 0; h2 < 2; ++h2) {
                vbf16x16 bfrag = cat8(kp[h2 * 4 + hi], kp[h2 * 4 + 2 + hi]);
                sacc = __builtin_amdgcn_wmma_f32_16x16x32_bf16(
                    false, aq[h2], false, bfrag, (short)0, sacc, false, false);
            }
            #pragma unroll
            for (int v = 0; v < 8; ++v) {
                int rq   = qb + hi * 8 + v;
                int kpos = kstart + sc * 16 + col;
                float sv = sacc[v] * 0.125f;
                if (kpos > rq) sv = -1e30f;
                pv[sc][v] = sv;
                chmax[v]  = fmaxf(chmax[v], sv);
            }
        }
        // ---- online softmax (row reductions across 16 lanes) ----
        #pragma unroll
        for (int v = 0; v < 8; ++v) {
            float x2 = chmax[v];
            x2 = fmaxf(x2, __shfl_xor(x2, 1, 32));
            x2 = fmaxf(x2, __shfl_xor(x2, 2, 32));
            x2 = fmaxf(x2, __shfl_xor(x2, 4, 32));
            x2 = fmaxf(x2, __shfl_xor(x2, 8, 32));
            float mnew = fmaxf(mrow[v], x2);
            float corr = __expf(mrow[v] - mnew);
            mrow[v] = mnew;
            float rsum = 0.f;
            #pragma unroll
            for (int sc = 0; sc < 2; ++sc) {
                float p = __expf(pv[sc][v] - mnew);
                pv[sc][v] = p;
                rsum += p;
            }
            rsum += __shfl_xor(rsum, 1, 32);
            rsum += __shfl_xor(rsum, 2, 32);
            rsum += __shfl_xor(rsum, 4, 32);
            rsum += __shfl_xor(rsum, 8, 32);
            lrow[v] = lrow[v] * corr + rsum;
            #pragma unroll
            for (int nc = 0; nc < 4; ++nc) acc[nc][v] *= corr;
        }
        // ---- P: C-layout -> A-layout via per-wave LDS ----
        #pragma unroll
        for (int sc = 0; sc < 2; ++sc)
            #pragma unroll
            for (int v = 0; v < 8; ++v)
                Pl[w][hi * 8 + v][sc * 16 + col] = (__bf16)pv[sc][v];
        asm volatile("s_wait_dscnt 0" ::: "memory");
        const vbf16x8* pp = (const vbf16x8*)&Pl[w][col][0];
        vbf16x16 pA = cat8(pp[hi], pp[2 + hi]);
        // ---- acc += P (16x32) x V (32x64), V^T rows contiguous ----
        #pragma unroll
        for (int nc = 0; nc < 4; ++nc) {
            const vbf16x8* vp = (const vbf16x8*)&Vs[cur][nc * 16 + col][0];
            vbf16x16 bfrag = cat8(vp[hi], vp[2 + hi]);
            acc[nc] = __builtin_amdgcn_wmma_f32_16x16x32_bf16(
                false, pA, false, bfrag, (short)0, acc[nc], false, false);
        }
    }

    const int b = bh >> 4, h = bh & 15;
    #pragma unroll
    for (int v = 0; v < 8; ++v) {
        float inv = 1.f / lrow[v];
        int s = qb + hi * 8 + v;
        #pragma unroll
        for (int nc = 0; nc < 4; ++nc) {
            int d = nc * 16 + col;
            size_t off = (((size_t)b * S_ + s) * H_ + h) * DK_ + d;
            o[off] = (__bf16)(acc[nc][v] * inv);
        }
    }
}

// ---------------------------------------------------------------------------
// Kernel 3: output projection, bf16 attn [M,E] x wo -> fp32 out. A tile via TDM.
// Each wave computes 32x64 (2 A-frags reuse 4 B-frags).
// ---------------------------------------------------------------------------
__global__ __launch_bounds__(256) void oproj_kernel(
    const __bf16* __restrict__ a_, const float* __restrict__ wo,
    float* __restrict__ out)
{
    __shared__ alignas(16) __bf16 As[TBM][TBK + 8];
    __shared__ alignas(32) __bf16 WsB[4][32][16];

    const int tid  = threadIdx.x;
    const int w    = tid >> 5;
    const int lane = tid & 31;
    const int hi   = lane >> 4;
    const int col  = lane & 15;
    const int m0   = blockIdx.x * TBM;
    const int n0   = blockIdx.y * TBN;
    const int nloc = tid & 63;
    const int kg   = tid >> 6;

    vf32x8 acc[2][4] = {};

    for (int k0 = 0; k0 < E_; k0 += TBK) {
        __syncthreads();
#if HAVE_TDM
        if (w == 0)
            tdm_load_2d((unsigned)(uintptr_t)(void*)&As[0][0],
                        a_ + (size_t)m0 * E_ + k0,
                        32, TBM, E_, M_, (unsigned long long)E_, 1u, 3u, 3u);
#else
        #pragma unroll
        for (int i = 0; i < 4; ++i) {
            int idx = i * 256 + tid;
            int r = idx >> 2, c8 = idx & 3;
            *(vbf16x8*)&As[r][c8 * 8] =
                *(const vbf16x8*)(a_ + (size_t)(m0 + r) * E_ + k0 + c8 * 8);
        }
#endif
        {
            vbf16x8 c8;
            #pragma unroll
            for (int i2 = 0; i2 < 8; ++i2)
                c8[i2] = (__bf16)wo[(size_t)(k0 + kg * 8 + i2) * E_ + n0 + nloc];
            int dl = (nloc & 15) | ((kg >> 1) << 4);
            *(vbf16x8*)&WsB[nloc >> 4][dl][(kg & 1) * 8] = c8;
        }
#if HAVE_TDM
        if (w == 0) __builtin_amdgcn_s_wait_tensorcnt(0);
#endif
        __syncthreads();

        const vbf16x8* xp0 = (const vbf16x8*)&As[w * 32 + col][0];
        const vbf16x8* xp1 = (const vbf16x8*)&As[w * 32 + 16 + col][0];
        vbf16x16 a0 = cat8(xp0[2 * hi], xp0[2 * hi + 1]);
        vbf16x16 a1 = cat8(xp1[2 * hi], xp1[2 * hi + 1]);
        #pragma unroll
        for (int nc = 0; nc < 4; ++nc) {
            vbf16x16 bfrag = *(const vbf16x16*)&WsB[nc][lane][0];
            acc[0][nc] = __builtin_amdgcn_wmma_f32_16x16x32_bf16(
                false, a0, false, bfrag, (short)0, acc[0][nc], false, false);
            acc[1][nc] = __builtin_amdgcn_wmma_f32_16x16x32_bf16(
                false, a1, false, bfrag, (short)0, acc[1][nc], false, false);
        }
    }

    #pragma unroll
    for (int mg = 0; mg < 2; ++mg)
        #pragma unroll
        for (int nc = 0; nc < 4; ++nc)
            #pragma unroll
            for (int v = 0; v < 8; ++v) {
                int m = m0 + w * 32 + mg * 16 + hi * 8 + v;
                int n = n0 + nc * 16 + col;
                out[(size_t)m * E_ + n] = acc[mg][nc][v];
            }
}

// ---------------------------------------------------------------------------
extern "C" void kernel_launch(void* const* d_in, const int* in_sizes, int n_in,
                              void* d_out, int out_size, void* d_ws, size_t ws_size,
                              hipStream_t stream) {
    const float* x  = (const float*)d_in[0];
    const float* wq = (const float*)d_in[1];
    const float* wk = (const float*)d_in[2];
    const float* wv = (const float*)d_in[3];
    const float* wo = (const float*)d_in[4];
    float* out = (float*)d_out;

    const size_t N = (size_t)M_ * E_;
    __bf16* qb  = (__bf16*)d_ws;
    __bf16* kb  = qb + N;
    __bf16* vtb = kb + N;    // transposed V [B,H,DK,S]
    __bf16* ab  = vtb + N;

    dim3 g1(M_ / TBM, E_ / TBN, 3);
    qkv_rope_kernel<<<g1, 256, 0, stream>>>(x, wq, wk, wv, qb, kb, vtb);

    dim3 g2(S_ / 128, B_ * H_);
    attn_kernel<<<g2, 256, 0, stream>>>(qb, kb, vtb, ab);

    dim3 g3(M_ / TBM, E_ / TBN);
    oproj_kernel<<<g3, 256, 0, stream>>>(ab, wo, out);
}
